// Organelle_Thalamus_83519934038252
// MI455X (gfx1250) — matrix operand order repaired and checked by reading, not compiled, not run on hardware
//
#include <hip/hip_runtime.h>
#include <math.h>

// MI455X / gfx1250, wave32. WMMA bf16 16x16x32 + async global->LDS staging.

typedef __bf16 v16bf __attribute__((ext_vector_type(16)));
typedef float  v8f   __attribute__((ext_vector_type(8)));

#define DIM_V   768
#define DIM_T   768
#define HID     256
#define NKEEP   96
#define BATCH   64
#define NPATCH  1024
#define SEQ     64

#define ROWS    64      // rows (patches) per block slab
#define ASTRIDE 776     // bf16 LDS row stride (768 + 8 pad -> lanes land 4 banks apart)
#define KSTEPS  (DIM_V / 32)

// ---------------------------------------------------------------------------
// Kernel 0: repack W1[0:768,0:256] fp32 -> bf16 in WMMA B-fragment order.
// Layout: wpk[((ks*HID + col) * 32) + t]  with t = K-within-step (0..31).
// A lane (col = lane%16, half-split at K+16) then reads one contiguous,
// 32B-aligned 32-byte chunk per k-step: 2x global_load_b128, no converts.
// ---------------------------------------------------------------------------
__global__ void __launch_bounds__(256)
ot_pack_w1(const float* __restrict__ W1, __bf16* __restrict__ wpk) {
  const int i = blockIdx.x * 256 + threadIdx.x;   // grid covers 768*256
  const int t    = i & 31;
  const int col  = (i >> 5) & (HID - 1);
  const int ks   = i >> 13;
  wpk[i] = (__bf16)W1[(size_t)(ks * 32 + t) * HID + col];
}

// ---------------------------------------------------------------------------
// Kernel 1: per-batch context bias  ctx_bias[b][h] = b1[h] + mean_s(text)·W1[768+d][h]
// ---------------------------------------------------------------------------
__global__ void __launch_bounds__(256)
ot_ctx_bias(const float* __restrict__ text, const float* __restrict__ W1,
            const float* __restrict__ b1, float* __restrict__ ctx_bias) {
  __shared__ float s_tm[DIM_T];
  const int b = blockIdx.x, tid = threadIdx.x;
  for (int d = tid; d < DIM_T; d += 256) {
    float s = 0.f;
    const float* p = text + (size_t)b * SEQ * DIM_T + d;
    #pragma unroll 4
    for (int si = 0; si < SEQ; ++si) s += p[si * DIM_T];
    s_tm[d] = s * (1.0f / SEQ);
  }
  __syncthreads();
  const int h = tid;                      // 256 threads == HID
  float acc = b1[h];
  for (int d = 0; d < DIM_T; ++d)
    acc += s_tm[d] * W1[(size_t)(DIM_V + d) * HID + h];
  ctx_bias[b * HID + h] = acc;
}

// ---------------------------------------------------------------------------
// Kernel 2: scores = sigmoid(relu(x·W1v + ctx_bias)·W2 + b2)  via WMMA bf16
// grid = BATCH * (NPATCH/ROWS) blocks, 512 threads (16 waves; wave w = col tile w)
// A slab staged with GLOBAL_LOAD_ASYNC_TO_LDS_B128 (ASYNCcnt), then one
// LDS->LDS fp32->bf16 convert pass.
// ---------------------------------------------------------------------------
__global__ void __launch_bounds__(512)
ot_score_wmma(const float* __restrict__ dv, const __bf16* __restrict__ wpk,
              const float* __restrict__ W2, const float* __restrict__ b2,
              const float* __restrict__ ctx_bias, float* __restrict__ scores) {
  extern __shared__ char smem[];
  float*  a_raw = (float*)smem;                                  // ROWS*DIM_V fp32 (192KB)
  __bf16* a_lds = (__bf16*)(smem + (size_t)ROWS * DIM_V * 4);    // ROWS*ASTRIDE bf16 (97KB)
  __shared__ float s_score[ROWS];

  const int tid  = threadIdx.x;
  const int wave = tid >> 5;
  const int lane = tid & 31;
  const int l16  = lane & 15;
  const int b    = blockIdx.x >> 4;            // NPATCH/ROWS == 16 slabs/batch
  const int row0 = (blockIdx.x & 15) * ROWS;
  const int col  = wave * 16 + l16;            // hidden-unit column owned by this lane

  // ---- Async stage: raw fp32 slab straight into LDS, no VGPR round trip.
  const float* abase = dv + ((size_t)b * NPATCH + row0) * DIM_V;
  for (int i = tid * 4; i < ROWS * DIM_V; i += 512 * 4) {
    const unsigned ldsoff = (unsigned)(uintptr_t)(a_raw + i);    // low 32b of generic LDS ptr
    const float*   g      = abase + i;                           // 16B-aligned
    asm volatile("global_load_async_to_lds_b128 %0, %1, off"
                 :: "v"(ldsoff), "v"(g) : "memory");
  }
  if (tid < ROWS) s_score[tid] = 0.f;

  // C init while copies are in flight: ctx_bias is column-constant -> fold into C.
  const float cb  = ctx_bias[b * HID + col];
  const float w2c = W2[col];
  v8f acc[4];
  #pragma unroll
  for (int rt = 0; rt < 4; ++rt)
    #pragma unroll
    for (int e = 0; e < 8; ++e) acc[rt][e] = cb;

  asm volatile("s_wait_asynccnt 0x0" ::: "memory");
  __syncthreads();

  // ---- One-shot convert pass fp32(LDS) -> bf16(LDS), padded stride.
  for (int i = tid; i < ROWS * DIM_V; i += 512) {
    const int r = i / DIM_V, k = i - r * DIM_V;
    a_lds[r * ASTRIDE + k] = (__bf16)a_raw[i];
  }
  __syncthreads();

  const int akoff = (lane < 16) ? 0 : 16;      // A layout: lanes 0-15 K..K+15, 16-31 K+16..31
  const int bkoff = (lane < 16) ? 0 : 16;      // B layout: lanes 0-15 K..K+15, 16-31 K+16..31

  for (int ks = 0; ks < KSTEPS; ++ks) {        // 24 k-steps of 32
    const int k0 = ks * 32;
    // Pre-packed bf16 B fragment: one contiguous 32B (L2-resident) read.
    const v16bf bfrag =
        *(const v16bf*)(wpk + ((size_t)(ks * HID + col)) * 32 + bkoff);

    #pragma unroll
    for (int rt = 0; rt < 4; ++rt) {
      const __bf16* ap = &a_lds[(rt * 16 + l16) * ASTRIDE + k0 + akoff];
      v16bf afrag;
      #pragma unroll
      for (int i = 0; i < 8; ++i) { afrag[i] = ap[i]; afrag[8 + i] = ap[8 + i]; }
      acc[rt] = __builtin_amdgcn_wmma_f32_16x16x32_bf16(
          false, afrag, false, bfrag, (short)0, acc[rt], false, false);
    }
  }

  // Fused epilogue: relu, x W2[col], reduce over this wave's 16 columns,
  // then cross-wave accumulate per row in LDS.
  #pragma unroll
  for (int rt = 0; rt < 4; ++rt) {
    #pragma unroll
    for (int r = 0; r < 8; ++r) {
      float hv = acc[rt][r];
      hv = hv > 0.f ? hv : 0.f;
      float p = hv * w2c;
      p += __shfl_xor(p, 1, 32);
      p += __shfl_xor(p, 2, 32);
      p += __shfl_xor(p, 4, 32);
      p += __shfl_xor(p, 8, 32);               // full sum within each 16-lane half
      if (l16 == 0) {
        const int row = rt * 16 + r + ((lane < 16) ? 0 : 8);  // C/D: VGPR r -> M=r / M=r+8
        atomicAdd(&s_score[row], p);
      }
    }
  }
  __syncthreads();
  if (tid < ROWS) {
    const float x = s_score[tid] + b2[0];
    scores[b * NPATCH + row0 + tid] = 1.0f / (1.0f + __expf(-x));
  }
}

// ---------------------------------------------------------------------------
// Kernel 3: per-batch top-96 (iterative argmax, low-index tie-break) + rank sort
// ---------------------------------------------------------------------------
__global__ void __launch_bounds__(256)
ot_topk(const float* __restrict__ scores, int* __restrict__ topk) {
  __shared__ float s[NPATCH];
  __shared__ float rv[256];
  __shared__ int   ri[256];
  __shared__ int   sel[NKEEP];
  __shared__ int   sorted[NKEEP];
  const int b = blockIdx.x, tid = threadIdx.x;

  for (int i = tid; i < NPATCH; i += 256) s[i] = scores[b * NPATCH + i];
  __syncthreads();

  for (int it = 0; it < NKEEP; ++it) {
    float mv = -1e30f; int mi = 0x7fffffff;
    for (int i = tid; i < NPATCH; i += 256) {
      const float v = s[i];
      if (v > mv || (v == mv && i < mi)) { mv = v; mi = i; }
    }
    rv[tid] = mv; ri[tid] = mi;
    __syncthreads();
    for (int off = 128; off > 0; off >>= 1) {
      if (tid < off) {
        const float ov = rv[tid + off]; const int oi = ri[tid + off];
        if (ov > rv[tid] || (ov == rv[tid] && oi < ri[tid])) { rv[tid] = ov; ri[tid] = oi; }
      }
      __syncthreads();
    }
    if (tid == 0) { sel[it] = ri[0]; s[ri[0]] = -1e38f; }
    __syncthreads();
  }

  if (tid < NKEEP) {                       // indices are distinct -> rank sort
    const int my = sel[tid];
    int rank = 0;
    for (int j = 0; j < NKEEP; ++j) rank += (sel[j] < my);
    sorted[rank] = my;
  }
  __syncthreads();
  if (tid < NKEEP) topk[b * NKEEP + tid] = sorted[tid];
}

// ---------------------------------------------------------------------------
// Kernel 4: gather kept rows + register token; emit indices. One block per row.
// ---------------------------------------------------------------------------
__global__ void __launch_bounds__(192)
ot_gather(const float* __restrict__ dv, const float* __restrict__ reg,
          const int* __restrict__ topk, float* __restrict__ fv,
          int* __restrict__ idx_out) {
  const int b = blockIdx.x / (NKEEP + 1);
  const int j = blockIdx.x - b * (NKEEP + 1);
  const float* src;
  if (j < NKEEP) {
    const int idx = topk[b * NKEEP + j];
    src = dv + ((size_t)b * NPATCH + idx) * DIM_V;
    if (threadIdx.x == 0) idx_out[b * NKEEP + j] = idx;
  } else {
    src = reg;
  }
  float4*       d4 = (float4*)(fv + ((size_t)b * (NKEEP + 1) + j) * DIM_V);
  const float4* s4 = (const float4*)src;
  d4[threadIdx.x] = s4[threadIdx.x];       // 192 threads * 16B = 768 floats
}

// ---------------------------------------------------------------------------
extern "C" void kernel_launch(void* const* d_in, const int* in_sizes, int n_in,
                              void* d_out, int out_size, void* d_ws, size_t ws_size,
                              hipStream_t stream) {
  const float* dv   = (const float*)d_in[0];   // [64,1024,768]
  const float* text = (const float*)d_in[1];   // [64,64,768]
  const float* W1   = (const float*)d_in[2];   // [1536,256]
  const float* b1   = (const float*)d_in[3];   // [256]
  const float* W2   = (const float*)d_in[4];   // [256,1]
  const float* b2   = (const float*)d_in[5];   // [1]
  const float* reg  = (const float*)d_in[6];   // [1,1,768]

  char*  ws       = (char*)d_ws;
  float* ctx_bias = (float*)ws;                               // 64*256  f32   (64 KB)
  float* scores   = (float*)(ws + 65536);                     // 64*1024 f32   (256 KB)
  int*   topk     = (int*)(ws + 65536 + 262144);              // 64*96   i32   (24 KB)
  __bf16* wpk     = (__bf16*)(ws + 65536 + 262144 + 24576);   // 768*256 bf16  (384 KB)

  float* fv      = (float*)d_out;                                        // [64,97,768]
  int*   idx_out = (int*)(fv + (size_t)BATCH * (NKEEP + 1) * DIM_V);     // [64,96]

  ot_pack_w1<<<(DIM_V * HID) / 256, 256, 0, stream>>>(W1, wpk);
  ot_ctx_bias<<<BATCH, 256, 0, stream>>>(text, W1, b1, ctx_bias);

  const size_t shmem = (size_t)ROWS * DIM_V * sizeof(float)      // raw fp32 slab
                     + (size_t)ROWS * ASTRIDE * sizeof(__bf16);  // bf16 slab (~289 KB)
  ot_score_wmma<<<BATCH * (NPATCH / ROWS), 512, shmem, stream>>>(
      dv, wpk, W2, b2, ctx_bias, scores);

  ot_topk<<<BATCH, 256, 0, stream>>>(scores, topk);

  ot_gather<<<BATCH * (NKEEP + 1), 192, 0, stream>>>(dv, reg, topk, fv, idx_out);
}
